// GNN_26130581029656
// MI455X (gfx1250) — matrix operand order repaired and checked by reading, not compiled, not run on hardware
//
#include <hip/hip_runtime.h>
#include <hip/hip_bf16.h>
#include <math.h>

typedef __attribute__((ext_vector_type(16))) _Float16 v16h;
typedef __attribute__((ext_vector_type(8)))  _Float16 v8h;
typedef __attribute__((ext_vector_type(8)))  float    v8f;
typedef __attribute__((ext_vector_type(4)))  float    v4f;

#define CH 128
#define WT_STRIDE 136   // padded LDS row stride (halves): 272 B -> conflict-free b128 reads
#define NEG_SLOPE 0.2f

// ---------------------------------------------------------------------------
// GEMM: Y[n][128] = X[n][128] @ W[128][128], f16 WMMA with f32 accumulate.
// Block = 256 threads = 8 waves; each stripe covers 128 rows (16 per wave).
// W is staged transposed (Wt[n][k], f16) so B fragments are two ds_load_b128
// (hoisted out of the stripe loop by the compiler into high VGPRs).
// A fragments come straight from global as float4 pairs, row index clamped so
// loads are unguarded; stores use a wave-uniform full-stripe fast path.
// ---------------------------------------------------------------------------
__global__ __launch_bounds__(256) void gemm128_wmma(
    const float* __restrict__ X, const float* __restrict__ W,
    float* __restrict__ Y, int nrows)
{
  __shared__ _Float16 Wt[CH * WT_STRIDE];   // 34 KB, transposed [n][k]

  const int tid  = threadIdx.x;
  const int wave = __builtin_amdgcn_readfirstlane(tid >> 5);  // scalar wave id
  const int lane = tid & 31;

  // Stage weights transposed (f32 -> f16); global reads coalesced
  for (int i = tid; i < CH * CH; i += 256) {
    int k = i >> 7, n = i & (CH - 1);
    Wt[n * WT_STRIDE + k] = (_Float16)W[i];
  }
  __syncthreads();

  const int mloc = lane & 15;
  const int kg   = (lane >> 4) * 8;        // A-layout K sub-offset per lane half
  const int nloc = lane & 15;
  const int kbB  = (lane >> 4) * 8;        // B-layout K offset per lane half
  const int rowOff = (lane >> 4) * 8;      // C/D layout: hi lanes hold M=r+8

  const int nStripes = (nrows + 127) >> 7;
  for (int stripe = blockIdx.x; stripe < nStripes; stripe += gridDim.x) {
    const int rowBase = stripe * 128 + wave * 16;        // SGPR
    const bool full = (rowBase + 16 <= nrows);           // wave-uniform
    int arow = rowBase + mloc;
    if (arow > nrows - 1) arow = nrows - 1;              // clamp: tail rows discarded at store

    // A fragments: per lane, two contiguous 8-float runs per K-tile (unguarded)
    v16h A[4];
    #pragma unroll
    for (int t = 0; t < 4; ++t) {
      const float* p = X + (long)arow * CH + t * 32 + kg;
      v4f a0 = *(const v4f*)(p);
      v4f a1 = *(const v4f*)(p + 4);
      v4f a2 = *(const v4f*)(p + 16);
      v4f a3 = *(const v4f*)(p + 20);
      #pragma unroll
      for (int e = 0; e < 4; ++e) {
        A[t][e]      = (_Float16)a0[e];
        A[t][4 + e]  = (_Float16)a1[e];
        A[t][8 + e]  = (_Float16)a2[e];
        A[t][12 + e] = (_Float16)a3[e];
      }
    }

    #pragma unroll
    for (int j = 0; j < 8; ++j) {          // 8 N-tiles of 16
      v8f acc = {};
      #pragma unroll
      for (int t = 0; t < 4; ++t) {        // 4 K-tiles of 32
        const _Float16* wp = &Wt[(j * 16 + nloc) * WT_STRIDE + t * 32 + kbB];
        v8h blo = *(const v8h*)(wp);
        v8h bhi = *(const v8h*)(wp + 16);
        v16h B = __builtin_shufflevector(blo, bhi,
                 0, 1, 2, 3, 4, 5, 6, 7, 8, 9, 10, 11, 12, 13, 14, 15);
        acc = __builtin_amdgcn_wmma_f32_16x16x32_f16(
            false, A[t], false, B, (short)0, acc, false, false);
      }
      float* yp = Y + (long)(rowBase + rowOff) * CH + j * 16 + nloc;
      if (full) {
        #pragma unroll
        for (int r = 0; r < 8; ++r) yp[(long)r * CH] = acc[r];
      } else {
        #pragma unroll
        for (int r = 0; r < 8; ++r)
          if (rowBase + rowOff + r < nrows) yp[(long)r * CH] = acc[r];
      }
    }
  }
}

// ---------------------------------------------------------------------------
// Elementwise / per-node / per-edge helpers
// ---------------------------------------------------------------------------
__device__ inline float lrelu(float x) { return x > 0.f ? x : NEG_SLOPE * x; }

__device__ inline void atomicMaxF(float* addr, float val) {
  if (val >= 0.f) atomicMax((int*)addr, __float_as_int(val));
  else            atomicMin((unsigned int*)addr, __float_as_uint(val));
}

__global__ void k_fill(float* p, float v, long n) {
  long i = (long)blockIdx.x * blockDim.x + threadIdx.x;
  for (; i < n; i += (long)gridDim.x * blockDim.x) p[i] = v;
}

__global__ void k_deg_count(const int* __restrict__ dst, float* __restrict__ deg, int E) {
  int i = blockIdx.x * blockDim.x + threadIdx.x;
  for (; i < E; i += gridDim.x * blockDim.x) atomicAdd(&deg[dst[i]], 1.0f);
}

__global__ void k_dinv(float* d, int n) {
  int i = blockIdx.x * blockDim.x + threadIdx.x;
  if (i < n) d[i] = rsqrtf(fmaxf(d[i], 1.0f));
}

// out[i][c] = h[i][c] * dinv[i]^2   (self-loop contribution)
__global__ void k_gcn_init(const float* __restrict__ h, const float* __restrict__ dinv,
                           float* __restrict__ out, int n) {
  int w = (blockIdx.x * blockDim.x + threadIdx.x) >> 5;
  int lane = threadIdx.x & 31;
  if (w >= n) return;
  float s = dinv[w] * dinv[w];
  #pragma unroll
  for (int c = lane; c < CH; c += 32) out[(long)w * CH + c] = h[(long)w * CH + c] * s;
}

__global__ void k_gcn_edge(const int* __restrict__ src, const int* __restrict__ dst,
                           const float* __restrict__ h, const float* __restrict__ dinv,
                           float* __restrict__ out, int E) {
  int w = (blockIdx.x * blockDim.x + threadIdx.x) >> 5;
  int lane = threadIdx.x & 31;
  int nw = (gridDim.x * blockDim.x) >> 5;
  for (int e = w; e < E; e += nw) {
    int s = src[e], d = dst[e];
    float norm = dinv[s] * dinv[d];
    const float* hs = h + (long)s * CH;
    float* od = out + (long)d * CH;
    #pragma unroll
    for (int c = lane; c < CH; c += 32) atomicAdd(&od[c], hs[c] * norm);
  }
}

__global__ void k_bias_relu(float* __restrict__ h, const float* __restrict__ b, long n) {
  long i = (long)blockIdx.x * blockDim.x + threadIdx.x;
  for (; i < n; i += (long)gridDim.x * blockDim.x)
    h[i] = fmaxf(h[i] + b[i & (CH - 1)], 0.0f);
}

// a_s[i] = h[i] . att_src ; a_d[i] = h[i] . att_dst   (wave per node)
__global__ void k_att(const float* __restrict__ h, const float* __restrict__ as_v,
                      const float* __restrict__ ad_v, float* __restrict__ a_s,
                      float* __restrict__ a_d, int n) {
  int w = (blockIdx.x * blockDim.x + threadIdx.x) >> 5;
  int lane = threadIdx.x & 31;
  if (w >= n) return;
  float ps = 0.f, pd = 0.f;
  #pragma unroll
  for (int c = lane; c < CH; c += 32) {
    float v = h[(long)w * CH + c];
    ps += v * as_v[c];
    pd += v * ad_v[c];
  }
  #pragma unroll
  for (int off = 16; off > 0; off >>= 1) {
    ps += __shfl_xor(ps, off, 32);
    pd += __shfl_xor(pd, off, 32);
  }
  if (lane == 0) { a_s[w] = ps; a_d[w] = pd; }
}

__global__ void k_gat_init_m(const float* __restrict__ a_s, const float* __restrict__ a_d,
                             float* __restrict__ mbuf, int n) {
  int i = blockIdx.x * blockDim.x + threadIdx.x;
  if (i < n) mbuf[i] = lrelu(a_s[i] + a_d[i]);   // self-loop edge always present
}

__global__ void k_gat_max(const int* __restrict__ src, const int* __restrict__ dst,
                          const float* __restrict__ a_s, const float* __restrict__ a_d,
                          float* __restrict__ mbuf, int E) {
  int i = blockIdx.x * blockDim.x + threadIdx.x;
  for (; i < E; i += gridDim.x * blockDim.x) {
    int s = src[i], d = dst[i];
    atomicMaxF(&mbuf[d], lrelu(a_s[s] + a_d[d]));
  }
}

__global__ void k_gat_denom_init(const float* __restrict__ a_s, const float* __restrict__ a_d,
                                 const float* __restrict__ mbuf, float* __restrict__ denom, int n) {
  int i = blockIdx.x * blockDim.x + threadIdx.x;
  if (i < n) denom[i] = expf(lrelu(a_s[i] + a_d[i]) - mbuf[i]);
}

__global__ void k_gat_denom(const int* __restrict__ src, const int* __restrict__ dst,
                            const float* __restrict__ a_s, const float* __restrict__ a_d,
                            const float* __restrict__ mbuf, float* __restrict__ denom, int E) {
  int i = blockIdx.x * blockDim.x + threadIdx.x;
  for (; i < E; i += gridDim.x * blockDim.x) {
    int s = src[i], d = dst[i];
    atomicAdd(&denom[d], expf(lrelu(a_s[s] + a_d[d]) - mbuf[d]));
  }
}

__global__ void k_gat_agg_init(const float* __restrict__ h, const float* __restrict__ a_s,
                               const float* __restrict__ a_d, const float* __restrict__ mbuf,
                               const float* __restrict__ denom, float* __restrict__ out, int n) {
  int w = (blockIdx.x * blockDim.x + threadIdx.x) >> 5;
  int lane = threadIdx.x & 31;
  if (w >= n) return;
  float alpha = expf(lrelu(a_s[w] + a_d[w]) - mbuf[w]) / denom[w];
  #pragma unroll
  for (int c = lane; c < CH; c += 32) out[(long)w * CH + c] = h[(long)w * CH + c] * alpha;
}

__global__ void k_gat_agg(const int* __restrict__ src, const int* __restrict__ dst,
                          const float* __restrict__ h, const float* __restrict__ a_s,
                          const float* __restrict__ a_d, const float* __restrict__ mbuf,
                          const float* __restrict__ denom, float* __restrict__ out, int E) {
  int w = (blockIdx.x * blockDim.x + threadIdx.x) >> 5;
  int lane = threadIdx.x & 31;
  int nw = (gridDim.x * blockDim.x) >> 5;
  for (int e = w; e < E; e += nw) {
    int s = src[e], d = dst[e];
    float alpha = expf(lrelu(a_s[s] + a_d[d]) - mbuf[d]) / denom[d];
    const float* hs = h + (long)s * CH;
    float* od = out + (long)d * CH;
    #pragma unroll
    for (int c = lane; c < CH; c += 32) atomicAdd(&od[c], hs[c] * alpha);
  }
}

__global__ void k_pool(const float* __restrict__ h, const int* __restrict__ batch,
                       float* __restrict__ sums, float* __restrict__ cnt, int n) {
  int w = (blockIdx.x * blockDim.x + threadIdx.x) >> 5;
  int lane = threadIdx.x & 31;
  if (w >= n) return;
  int g = batch[w];
  #pragma unroll
  for (int c = lane; c < CH; c += 32)
    atomicAdd(&sums[(long)g * CH + c], h[(long)w * CH + c]);
  if (lane == 0) atomicAdd(&cnt[g], 1.0f);
}

__global__ void k_final(const float* __restrict__ sums, const float* __restrict__ cnt,
                        const float* __restrict__ fcW, const float* __restrict__ fcb,
                        float* __restrict__ out) {
  int idx = blockIdx.x * blockDim.x + threadIdx.x;   // 128 graphs * 64 outputs
  if (idx >= 128 * 64) return;
  int g = idx >> 6, o = idx & 63;
  float c = fmaxf(cnt[g], 1.0f);
  float acc = 0.f;
  #pragma unroll 4
  for (int k = 0; k < CH; ++k) acc += sums[g * CH + k] * fcW[k * 64 + o];
  out[idx] = acc / c + fcb[o];
}

// ---------------------------------------------------------------------------
extern "C" void kernel_launch(void* const* d_in, const int* in_sizes, int n_in,
                              void* d_out, int out_size, void* d_ws, size_t ws_size,
                              hipStream_t stream) {
  const float* x       = (const float*)d_in[0];
  const int*   edge    = (const int*)d_in[1];
  const int*   batch   = (const int*)d_in[2];
  const float* W1      = (const float*)d_in[3];
  const float* b1      = (const float*)d_in[4];
  const float* W2      = (const float*)d_in[5];
  const float* att_src = (const float*)d_in[6];
  const float* att_dst = (const float*)d_in[7];
  const float* b2      = (const float*)d_in[8];
  const float* fcW     = (const float*)d_in[9];
  const float* fcb     = (const float*)d_in[10];
  float* out = (float*)d_out;

  const int N = in_sizes[0] / CH;
  const int E = in_sizes[1] / 2;
  const int* esrc = edge;
  const int* edst = edge + E;

  // Workspace carve-up (~105 MB): two NxCH buffers + per-node scalars + pool
  float* bufA  = (float*)d_ws;           // h1, then h2
  float* bufB  = bufA + (long)N * CH;    // GCN out/g, then GAT out
  float* dinv  = bufB + (long)N * CH;
  float* a_s   = dinv + N;
  float* a_d   = a_s + N;
  float* mbuf  = a_d + N;
  float* denom = mbuf + N;
  float* sums  = denom + N;              // 128*128
  float* cnt   = sums + 128 * 128;       // 128

  const int gemmBlocks = 512;                      // grid-stride over 128-row stripes
  const int nodeWaveBlocks = (N + 7) / 8;          // 8 waves / 256-thread block
  const int elemBlocks = 4096;
  const int edgeBlocks = 4096;

  // ---- GCN layer ----
  gemm128_wmma<<<gemmBlocks, 256, 0, stream>>>(x, W1, bufA, N);          // h1 = x@W1
  k_fill<<<512, 256, 0, stream>>>(dinv, 1.0f, N);                        // deg = 1 (self loop)
  k_deg_count<<<edgeBlocks, 256, 0, stream>>>(edst, dinv, E);
  k_dinv<<<(N + 255) / 256, 256, 0, stream>>>(dinv, N);
  k_gcn_init<<<nodeWaveBlocks, 256, 0, stream>>>(bufA, dinv, bufB, N);
  k_gcn_edge<<<edgeBlocks, 256, 0, stream>>>(esrc, edst, bufA, dinv, bufB, E);
  k_bias_relu<<<elemBlocks, 256, 0, stream>>>(bufB, b1, (long)N * CH);   // g

  // ---- GAT layer ----
  gemm128_wmma<<<gemmBlocks, 256, 0, stream>>>(bufB, W2, bufA, N);       // h2 = g@W2
  k_att<<<nodeWaveBlocks, 256, 0, stream>>>(bufA, att_src, att_dst, a_s, a_d, N);
  k_gat_init_m<<<(N + 255) / 256, 256, 0, stream>>>(a_s, a_d, mbuf, N);
  k_gat_max<<<edgeBlocks, 256, 0, stream>>>(esrc, edst, a_s, a_d, mbuf, E);
  k_gat_denom_init<<<(N + 255) / 256, 256, 0, stream>>>(a_s, a_d, mbuf, denom, N);
  k_gat_denom<<<edgeBlocks, 256, 0, stream>>>(esrc, edst, a_s, a_d, mbuf, denom, E);
  k_gat_agg_init<<<nodeWaveBlocks, 256, 0, stream>>>(bufA, a_s, a_d, mbuf, denom, bufB, N);
  k_gat_agg<<<edgeBlocks, 256, 0, stream>>>(esrc, edst, bufA, a_s, a_d, mbuf, denom, bufB, E);
  k_bias_relu<<<elemBlocks, 256, 0, stream>>>(bufB, b2, (long)N * CH);

  // ---- Pool + FC ----
  k_fill<<<64, 256, 0, stream>>>(sums, 0.0f, 128 * 128 + 128);           // sums + cnt contiguous
  k_pool<<<nodeWaveBlocks, 256, 0, stream>>>(bufB, batch, sums, cnt, N);
  k_final<<<(128 * 64 + 255) / 256, 256, 0, stream>>>(sums, cnt, fcW, fcb, out);
}
